// GNNLayerVariableEncoder_88502096101408
// MI455X (gfx1250) — matrix operand order repaired and checked by reading, not compiled, not run on hardware
//
#include <hip/hip_runtime.h>

#define Bn   1024
#define Ln   16
#define Tn   (Bn * Ln)     // 16384 packed rows (static bound)
#define PWn  1024
#define PBn  32
#define KDIM 1056          // D*D + D = 1024 + 32, divisible by 32
#define NDIM 2048          // HID
#define MB   64
#define NB   128
#define KB   32
#define NK   (KDIM / KB)   // 33 K-steps

typedef __attribute__((ext_vector_type(16))) _Float16 v16h;
typedef __attribute__((ext_vector_type(8)))  float    v8f;
typedef int v4i_gcc __attribute__((vector_size(16)));   // matches builtin proto

// ---------------------------------------------------------------------------
// CDNA5 async global->LDS (16B per lane, tracked by ASYNCcnt, no VGPR bounce)
// ---------------------------------------------------------------------------
#if __has_builtin(__builtin_amdgcn_global_load_async_to_lds_b128)
__device__ __forceinline__ void async_load_b128(const void* g, void* l)
{
    __builtin_amdgcn_global_load_async_to_lds_b128(
        (__attribute__((address_space(1))) v4i_gcc*)g,
        (__attribute__((address_space(3))) v4i_gcc*)l, 0, 0);
}
#else
__device__ __forceinline__ void async_load_b128(const void* g, void* l)
{
    const unsigned lds = (unsigned)(unsigned long long)(uintptr_t)l; // LDS off = addr[31:0]
    const unsigned long long ga = (unsigned long long)(uintptr_t)g;
    asm volatile("global_load_async_to_lds_b128 %0, %1, off"
                 :: "v"(lds), "v"(ga) : "memory");
}
#endif

__device__ __forceinline__ void wait_async3()
{
#if __has_builtin(__builtin_amdgcn_s_wait_asynccnt)
    __builtin_amdgcn_s_wait_asynccnt(3);
#else
    asm volatile("s_wait_asynccnt 0x3" ::: "memory");
#endif
}

__device__ __forceinline__ void wait_async0()
{
#if __has_builtin(__builtin_amdgcn_s_wait_asynccnt)
    __builtin_amdgcn_s_wait_asynccnt(0);
#else
    asm volatile("s_wait_asynccnt 0x0" ::: "memory");
#endif
}

// ---------------------------------------------------------------------------
// Kernel 1: replicate pack_padded_sequence ordering + repeat_interleave plens.
// ---------------------------------------------------------------------------
__global__ void pack_kernel(const int* __restrict__ gnn_layers,
                            const int* __restrict__ wparams,
                            const int* __restrict__ bparams,
                            int* __restrict__ src_b, int* __restrict__ src_t,
                            int* __restrict__ wpl,  int* __restrict__ bpl,
                            int* __restrict__ nvalid)
{
    __shared__ int len[Bn];
    __shared__ int sorted_idx[Bn];
    __shared__ int batch_sizes[Ln];
    __shared__ int offsets[Ln];
    __shared__ int cum[Bn];

    const int tid = threadIdx.x;   // blockDim.x == 1024
    len[tid] = gnn_layers[tid];
    __syncthreads();

    if (tid == 0) {
        int count[Ln + 2];
        for (int v = 0; v <= Ln + 1; ++v) count[v] = 0;
        for (int b = 0; b < Bn; ++b) count[len[b]]++;
        // stable descending counting sort
        int fill[Ln + 2];
        int acc = 0;
        for (int v = Ln; v >= 1; --v) { fill[v] = acc; acc += count[v]; }
        for (int b = 0; b < Bn; ++b) sorted_idx[fill[len[b]]++] = b;
        // batch_sizes[t] = #lengths > t ; offsets = exclusive prefix
        int off = 0;
        for (int t = 0; t < Ln; ++t) {
            int c = 0;
            for (int v = t + 1; v <= Ln; ++v) c += count[v];
            batch_sizes[t] = c;
            offsets[t] = off;
            off += c;
        }
        nvalid[0] = off;
        // exclusive prefix of lengths in ORIGINAL order (network-major plens)
        int c = 0;
        for (int b = 0; b < Bn; ++b) { cum[b] = c; c += len[b]; }
    }
    __syncthreads();

    for (int i = tid; i < Tn; i += 1024) {
        src_b[i] = 0; src_t[i] = 0; wpl[i] = 0; bpl[i] = 0;
    }
    __syncthreads();

    // scatter packed (time-major, sorted) gather indices
    for (int i = tid; i < Ln * Bn; i += 1024) {
        const int t = i >> 10, r = i & (Bn - 1);
        if (r < batch_sizes[t]) {
            const int p = offsets[t] + r;
            src_b[p] = sorted_idx[r];
            src_t[p] = t;
        }
    }
    // scatter plens (network-major, ORIGINAL index order — matches reference)
    {
        const int b = tid;
        const int l = len[b], c0 = cum[b];
        const int wv = wparams[b], bv = bparams[b];
        for (int j = 0; j < l; ++j) { wpl[c0 + j] = wv; bpl[c0 + j] = bv; }
    }
}

// ---------------------------------------------------------------------------
// Kernel 2: enc_w f32 -> f16 (layout [n][k] row-major, already B-friendly)
// ---------------------------------------------------------------------------
__global__ void convb_kernel(const float* __restrict__ src,
                             _Float16* __restrict__ dst, int n4)
{
    const int i = blockIdx.x * blockDim.x + threadIdx.x;
    if (i < n4) {
        const float4 f = ((const float4*)src)[i];
        _Float16 h[4] = { (_Float16)f.x, (_Float16)f.y, (_Float16)f.z, (_Float16)f.w };
        *(uint2*)(dst + (size_t)i * 4) = *(const uint2*)h;
    }
}

__device__ __forceinline__ float block_reduce(float v, float* red, int tid)
{
    red[tid] = v;
    __syncthreads();
    for (int s = 128; s > 0; s >>= 1) {
        if (tid < s) red[tid] += red[tid + s];
        __syncthreads();
    }
    const float r = red[0];
    __syncthreads();
    return r;
}

// ---------------------------------------------------------------------------
// Kernel 3: per packed row: masked sums -> 4 scalars -> f16 embedding row.
// ---------------------------------------------------------------------------
__global__ __launch_bounds__(256)
void emb_kernel(const float* __restrict__ weights, const float* __restrict__ biasv,
                const int* __restrict__ src_b, const int* __restrict__ src_t,
                const int* __restrict__ wpl,  const int* __restrict__ bpl,
                const float* __restrict__ w_tr_w, const float* __restrict__ w_tr_b,
                const float* __restrict__ b_tr_w, const float* __restrict__ b_tr_b,
                _Float16* __restrict__ Ah)
{
    __shared__ float red[256];
    const int p   = blockIdx.x;
    const int tid = threadIdx.x;
    const int sb = src_b[p], st = src_t[p];
    const int wl = wpl[p],  bl = bpl[p];

    const float* wrow = weights + ((size_t)sb * Ln + st) * PWn;
    const float4 w4 = ((const float4*)wrow)[tid];
    const int k0 = tid * 4;
    float s = 0.f;
    if (k0 + 0 < wl) s += w4.x;
    if (k0 + 1 < wl) s += w4.y;
    if (k0 + 2 < wl) s += w4.z;
    if (k0 + 3 < wl) s += w4.w;
    const float s_w = block_reduce(s, red, tid);

    float sbv = 0.f;
    if (tid < PBn) {
        const float bv = (biasv + ((size_t)sb * Ln + st) * PBn)[tid];
        sbv = (tid < bl) ? bv : 0.f;
    }
    const float s_b = block_reduce(sbv, red, tid);

    const float n_w = (float)wl, n_b = (float)bl;
    _Float16* arow = Ah + (size_t)p * KDIM;
    for (int k = tid; k < PWn; k += 256) {
        const float x = s_w * w_tr_w[k] + n_w * w_tr_b[k];
        arow[k] = (_Float16)(1.f / (1.f + __expf(-x)));
    }
    if (tid < PBn) {
        const float x = s_b * b_tr_w[tid] + n_b * b_tr_b[tid];
        arow[PWn + tid] = (_Float16)(1.f / (1.f + __expf(-x)));
    }
}

// ---------------------------------------------------------------------------
// Kernel 4: C[16384x2048] = relu(A[16384x1056] @ B^T + enc_b), masked rows.
// Triple-buffered async global->LDS pipeline; one barrier per K-step.
// Block tile 64x128, 8 waves, wave tile 32x32 = 2x2 v_wmma_f32_16x16x32_f16.
// ---------------------------------------------------------------------------
__global__ __launch_bounds__(256)
void gemm_kernel(const _Float16* __restrict__ Ah, const _Float16* __restrict__ Bh,
                 const float* __restrict__ enc_b, const int* __restrict__ nvalid_p,
                 float* __restrict__ out)
{
    __shared__ _Float16 As[3][MB * 40];   // 3 x 5120 B
    __shared__ _Float16 Bs[3][NB * 40];   // 3 x 10240 B

    const int tid    = threadIdx.x;
    const int wave   = tid >> 5;
    const int lane   = tid & 31;
    const int lane16 = lane & 15;
    const int hi     = lane >> 4;
    const int wm = wave >> 2;          // 0..1 -> M offset wm*32
    const int wn = wave & 3;           // 0..3 -> N offset wn*32
    const int mb = blockIdx.x * MB;
    const int nb = blockIdx.y * NB;

    // per-thread staging assignment (uniform, no divergence)
    const int arow = tid >> 2, ach = tid & 3;          // A: 64 rows x 4 chunks
    const int brow = tid >> 2, bch = tid & 3;          // B: first 64 n-rows
    const _Float16* agp = Ah + (size_t)(mb + arow) * KDIM + ach * 8;
    const _Float16* bgp0 = Bh + (size_t)(nb + brow)      * KDIM + bch * 8;
    const _Float16* bgp1 = Bh + (size_t)(nb + brow + 64) * KDIM + bch * 8;

    // issue 3 async-to-LDS ops (1 A chunk + 2 B chunks) for stage `buf`
    auto stage = [&](int buf, int ko) {
        async_load_b128(agp  + ko, &As[buf][arow * 40 + ach * 8]);
        async_load_b128(bgp0 + ko, &Bs[buf][brow * 40 + bch * 8]);
        async_load_b128(bgp1 + ko, &Bs[buf][(brow + 64) * 40 + bch * 8]);
    };

    v8f acc[2][2] = {};

    stage(0, 0);                                   // prologue: prefetch stage 0

    for (int it = 0; it < NK; ++it) {
        const int cur = it % 3;
        if (it + 1 < NK) {
            stage((it + 1) % 3, (it + 1) * KB);    // overlap next K-tile
            wait_async3();                         // oldest 3 (= stage `it`) done
        } else {
            wait_async0();                         // drain for last stage
        }
        __syncthreads();                           // all waves' stage `it` visible

        // A fragment: lane16 = M row; lanes 0-15 K=[hi*8..+7] (v0-3),
        // K=[16+hi*8..+7] (v4-7) per the 16-bit 16x32 A layout.
        v16h afrag[2], bfrag[2];
        for (int sm = 0; sm < 2; ++sm) {
            const int row = wm * 32 + sm * 16 + lane16;
            uint4* dst = (uint4*)&afrag[sm];
            dst[0] = *(const uint4*)(&As[cur][row * 40 + hi * 8]);
            dst[1] = *(const uint4*)(&As[cur][row * 40 + 16 + hi * 8]);
        }
        // B fragment: lane16 = N col; K=[hi*8..+7] (v0-3), K=[16+hi*8..+7] (v4-7), matching A.
        for (int sn = 0; sn < 2; ++sn) {
            const int nrow = wn * 32 + sn * 16 + lane16;
            uint4* dst = (uint4*)&bfrag[sn];
            dst[0] = *(const uint4*)(&Bs[cur][nrow * 40 + hi * 8]);
            dst[1] = *(const uint4*)(&Bs[cur][nrow * 40 + 16 + hi * 8]);
        }
        for (int sm = 0; sm < 2; ++sm)
            for (int sn = 0; sn < 2; ++sn)
                acc[sm][sn] = __builtin_amdgcn_wmma_f32_16x16x32_f16(
                    false, afrag[sm], false, bfrag[sn],
                    (short)0, acc[sm][sn], false, false);
        // no trailing barrier: buffer `cur` is rewritten 3 iterations later,
        // separated by the barrier at it+1 (triple buffering)
    }

    // fused epilogue: + enc_b, ReLU, zero rows >= n_valid
    const int nval = *nvalid_p;
    for (int sm = 0; sm < 2; ++sm) {
        for (int sn = 0; sn < 2; ++sn) {
            const int col = nb + wn * 32 + sn * 16 + lane16;
            const float eb = enc_b[col];
            for (int i = 0; i < 8; ++i) {
                const int row = mb + wm * 32 + sm * 16 + hi * 8 + i;  // f32 C layout
                float v = acc[sm][sn][i] + eb;
                v = v > 0.f ? v : 0.f;
                if (row >= nval) v = 0.f;
                out[(size_t)row * NDIM + col] = v;
            }
        }
    }
}

// ---------------------------------------------------------------------------
extern "C" void kernel_launch(void* const* d_in, const int* in_sizes, int n_in,
                              void* d_out, int out_size, void* d_ws, size_t ws_size,
                              hipStream_t stream)
{
    const float* weights = (const float*)d_in[0];   // [B, L, PW]
    const float* biasv   = (const float*)d_in[1];   // [B, L, PB]
    const int*   wparams = (const int*)d_in[2];     // [B]
    const int*   bparams = (const int*)d_in[3];     // [B]
    const int*   glayers = (const int*)d_in[4];     // [B]
    const float* w_tr_w  = (const float*)d_in[5];   // [1024]
    const float* w_tr_b  = (const float*)d_in[6];   // [1024]
    const float* b_tr_w  = (const float*)d_in[7];   // [32]
    const float* b_tr_b  = (const float*)d_in[8];   // [32]
    const float* enc_w   = (const float*)d_in[9];   // [2048, 1056]
    const float* enc_bv  = (const float*)d_in[10];  // [2048]
    float* out = (float*)d_out;                     // [T, 2048]

    char* ws = (char*)d_ws;
    int* src_b = (int*)ws;
    int* src_t = src_b + Tn;
    int* wpl   = src_t + Tn;
    int* bpl   = wpl + Tn;
    int* nval  = bpl + Tn;
    _Float16* Bh = (_Float16*)(ws + (size_t)4 * Tn * 4 + 256);  // 16B-aligned
    _Float16* Ah = Bh + (size_t)NDIM * KDIM;

    pack_kernel<<<1, 1024, 0, stream>>>(glayers, wparams, bparams,
                                        src_b, src_t, wpl, bpl, nval);

    const int n4 = (NDIM * KDIM) / 4;               // 540672
    convb_kernel<<<(n4 + 255) / 256, 256, 0, stream>>>(enc_w, Bh, n4);

    emb_kernel<<<Tn, 256, 0, stream>>>(weights, biasv, src_b, src_t, wpl, bpl,
                                       w_tr_w, w_tr_b, b_tr_w, b_tr_b, Ah);

    dim3 grid(Tn / MB, NDIM / NB);                  // 256 x 16
    gemm_kernel<<<grid, 256, 0, stream>>>(Ah, Bh, enc_bv, nval, out);
}
